// CrossAttention_3994319586123
// MI455X (gfx1250) — matrix operand order repaired and checked by reading, not compiled, not run on hardware
//
#include <hip/hip_runtime.h>
#include <hip/hip_bf16.h>
#include <float.h>
#include <stdint.h>

typedef __attribute__((ext_vector_type(16))) __bf16 v16bf;
typedef __attribute__((ext_vector_type(8)))  float  v8f;
typedef __bf16 bf16;

#define WMMA_BF16(a, b, c) \
  __builtin_amdgcn_wmma_f32_16x16x32_bf16(false, (a), false, (b), (short)0, (c), false, false)

__device__ inline v8f zero8() {
  v8f z;
#pragma unroll
  for (int i = 0; i < 8; ++i) z[i] = 0.f;
  return z;
}

// Async 16-byte global -> LDS copy (per active lane), tracked by ASYNCcnt.
__device__ inline void async_b128(void* lds_ptr, const void* gptr) {
  unsigned int lds_off = (unsigned int)(uintptr_t)lds_ptr;     // flat LDS addr bits [31:0] = LDS offset
  unsigned long long ga = (unsigned long long)(uintptr_t)gptr;
  asm volatile("global_load_async_to_lds_b128 %0, %1, off"
               :: "v"(lds_off), "v"(ga) : "memory");
}
__device__ inline void wait_async0() {
  asm volatile("s_wait_asynccnt 0" ::: "memory");
}

// A-fragment (16-bit, 16x32, MxK): lane&15 = row, (lane>>4)*8 = K interleave base.
__device__ inline v16bf load_a16(const bf16* base, int strideE, int rbase, int k0) {
  const int lane = threadIdx.x & 31;
  const int row = rbase + (lane & 15);
  const int kh = k0 + ((lane >> 4) << 3);
  union { v16bf v; uint4 q[2]; } u;
  u.q[0] = *(const uint4*)(base + row * strideE + kh);
  u.q[1] = *(const uint4*)(base + row * strideE + kh + 16);
  return u.v;
}

// B-fragment (32x16 KxN) from an N-major LDS tile: lane&15 = col, (lane>>4)*8 = K interleave base.
__device__ inline v16bf load_b16(const bf16* base, int strideE, int nbase, int k0) {
  const int lane = threadIdx.x & 31;
  const int n = nbase + (lane & 15);
  const int kh = k0 + ((lane >> 4) << 3);
  union { v16bf v; uint4 q[2]; } u;
  u.q[0] = *(const uint4*)(base + n * strideE + kh);
  u.q[1] = *(const uint4*)(base + n * strideE + kh + 16);
  return u.v;
}

// ---------------- preprocessing ----------------
__global__ __launch_bounds__(256) void f32_to_bf16_kernel(const float* __restrict__ in,
                                                          bf16* __restrict__ out, long long n4) {
  long long i = ((long long)blockIdx.x * 256 + threadIdx.x);
  if (i >= n4) return;
  float4 f = *(const float4*)(in + i * 4);
  union { unsigned long long u; bf16 h[4]; } p;
  p.h[0] = (bf16)f.x; p.h[1] = (bf16)f.y; p.h[2] = (bf16)f.z; p.h[3] = (bf16)f.w;
  *(unsigned long long*)(out + i * 4) = p.u;
}

// Wt[n][k] = (bf16)W[k][n]; K,N multiples of 32. Grid (K/32, N/32), 256 threads (32x8).
__global__ __launch_bounds__(256) void transpose_w_kernel(const float* __restrict__ W,
                                                          bf16* __restrict__ Wt, int K, int N) {
  __shared__ float t[32][33];
  const int k0 = blockIdx.x * 32, n0 = blockIdx.y * 32;
  const int tx = threadIdx.x & 31, ty = threadIdx.x >> 5;
#pragma unroll
  for (int i = ty; i < 32; i += 8) t[i][tx] = W[(size_t)(k0 + i) * N + n0 + tx];
  __syncthreads();
#pragma unroll
  for (int i = ty; i < 32; i += 8) Wt[(size_t)(n0 + i) * K + k0 + tx] = (bf16)t[tx][i];
}

// -------------------------------------------------------------------------
// Tiled WMMA GEMM, all-bf16 inputs, async double-buffered LDS pipeline.
// C[Mrows x N] = scale * (A[Mrows x K] @ Bt[N x K]^T) (+bias if OUT_F32)
// Block tile 128x128, K-step 32, 256 threads = 8 waves, each wave 32x64.
// -------------------------------------------------------------------------
template <bool OUT_F32>
__global__ __launch_bounds__(256) void gemm_wmma_kernel(
    const bf16* __restrict__ A, const bf16* __restrict__ Bt,
    void* __restrict__ Cp, const float* __restrict__ bias,
    int Mrows, int K, int N, float scale) {
  __shared__ bf16 lA[2][128][40];   // row stride 80 B (mult of 16)
  __shared__ bf16 lB[2][128][40];   // N-major B tile

  const int tid = threadIdx.x;
  const int n0 = blockIdx.x * 128;
  const int m0 = blockIdx.y * 128;
  const int lane = tid & 31;
  const int wave = tid >> 5;
  const int wm = (wave >> 1) * 32;
  const int wn = (wave & 1) * 64;
  const bool fullM = (m0 + 128 <= Mrows);

  auto fillA = [&](int buf, int k0) {
    if (fullM) {
#pragma unroll
      for (int t = 0; t < 2; ++t) {
        int c = tid + t * 256;                   // 512 chunks of 8 bf16
        int row = c >> 2, c8 = (c & 3) * 8;
        async_b128(&lA[buf][row][c8], A + (size_t)(m0 + row) * K + k0 + c8);
      }
    } else {
#pragma unroll
      for (int t = 0; t < 2; ++t) {
        int c = tid + t * 256;
        int row = c >> 2, c8 = (c & 3) * 8;
        uint4 val = make_uint4(0, 0, 0, 0);
        if (m0 + row < Mrows) val = *(const uint4*)(A + (size_t)(m0 + row) * K + k0 + c8);
        *(uint4*)&lA[buf][row][c8] = val;
      }
    }
  };
  auto fillB = [&](int buf, int k0) {
#pragma unroll
    for (int t = 0; t < 2; ++t) {
      int c = tid + t * 256;
      int row = c >> 2, c8 = (c & 3) * 8;
      async_b128(&lB[buf][row][c8], Bt + (size_t)(n0 + row) * K + k0 + c8);
    }
  };

  v8f acc[2][4];
#pragma unroll
  for (int i = 0; i < 2; ++i)
#pragma unroll
    for (int j = 0; j < 4; ++j) acc[i][j] = zero8();

  const int nk = K >> 5;
  fillA(0, 0);
  fillB(0, 0);
  for (int it = 0; it < nk; ++it) {
    const int buf = it & 1;
    wait_async0();            // my async fills of `buf` complete
    __syncthreads();          // everyone's fills done; everyone done reading buf^1
    if (it + 1 < nk) {        // prefetch next tile while computing this one
      fillA(buf ^ 1, (it + 1) * 32);
      fillB(buf ^ 1, (it + 1) * 32);
    }
    v16bf af[2], bfr[4];
#pragma unroll
    for (int i = 0; i < 2; ++i) af[i] = load_a16(&lA[buf][0][0], 40, wm + i * 16, 0);
#pragma unroll
    for (int j = 0; j < 4; ++j) bfr[j] = load_b16(&lB[buf][0][0], 40, wn + j * 16, 0);
#pragma unroll
    for (int i = 0; i < 2; ++i)
#pragma unroll
      for (int j = 0; j < 4; ++j) acc[i][j] = WMMA_BF16(af[i], bfr[j], acc[i][j]);
  }

  // Epilogue. C/D layout: lane holds col = lane&15, rows r + 8*(lane>>4).
  const int ln = lane & 15;
  const int h8 = (lane >> 4) * 8;
#pragma unroll
  for (int i = 0; i < 2; ++i) {
#pragma unroll
    for (int j = 0; j < 4; ++j) {
      const int col = n0 + wn + j * 16 + ln;
      if (OUT_F32) {
        float* C = (float*)Cp;
        const float bv = bias ? bias[col] : 0.f;
#pragma unroll
        for (int r = 0; r < 8; ++r) {
          const int row = m0 + wm + i * 16 + h8 + r;
          if (row < Mrows) C[(size_t)row * N + col] = acc[i][j][r] * scale + bv;
        }
      } else {
        bf16* C = (bf16*)Cp;
#pragma unroll
        for (int r = 0; r < 8; ++r) {
          const int row = m0 + wm + i * 16 + h8 + r;
          if (row < Mrows) C[(size_t)row * N + col] = (bf16)(acc[i][j][r] * scale);
        }
      }
    }
  }
}

// -------------------------------------------------------------------------
// Fused attention per (b, h, 64-row tile of N):
//   S = Qtile @ K^T (scale folded into q), masked softmax over M=77, O = P @ V.
// -------------------------------------------------------------------------
__global__ __launch_bounds__(256) void attn_kernel(
    const bf16* __restrict__ q, const bf16* __restrict__ kk,
    const bf16* __restrict__ vp, const unsigned char* __restrict__ mask,
    bf16* __restrict__ ao) {
  __shared__ bf16 lQ[64][72];    // 64x64 Q tile
  __shared__ bf16 lKt[80][72];   // K rows (B^T for S), rows 77..79 = 0
  __shared__ float lS[64][84];   // fp32 scores; later aliased as bf16 P (stride 168 bf16)
  __shared__ bf16 lVt[64][104];  // V transposed [d][m], m 77..95 = 0
  __shared__ float lAllow[80];

  const int tid = threadIdx.x;
  const int lane = tid & 31;
  const int wave = tid >> 5;
  const int n0 = blockIdx.x * 64;
  const int h = blockIdx.y;
  const int b = blockIdx.z;
  const size_t qrow0 = (size_t)b * 4096 + n0;
  const int hc = h * 64;

  // ---- stage Q async (always in-bounds), K/V/mask sync (guards/transpose) ----
#pragma unroll
  for (int t = 0; t < 2; ++t) {
    int c = tid + t * 256;
    int r = c >> 3, d8 = (c & 7) * 8;
    async_b128(&lQ[r][d8], q + (qrow0 + r) * 512 + hc + d8);
  }
  for (int c = tid; c < 640; c += 256) {
    int m = c >> 3, d8 = (c & 7) * 8;
    uint4 val = make_uint4(0, 0, 0, 0);
    if (m < 77) val = *(const uint4*)(kk + ((size_t)b * 77 + m) * 512 + hc + d8);
    *(uint4*)&lKt[m][d8] = val;
  }
  for (int c = tid; c < 64 * 96; c += 256) {
    int d = c / 96, m = c % 96;
    bf16 val = (bf16)0.f;
    if (m < 77) val = vp[((size_t)b * 77 + m) * 512 + hc + d];
    lVt[d][m] = val;
  }
  if (tid < 80) lAllow[tid] = (tid < 77 && mask[b * 77 + tid]) ? 1.f : 0.f;
  wait_async0();
  __syncthreads();

  // ---- S = Q @ K^T : 64x80 = 4x5 tiles ----
  for (int t = wave; t < 20; t += 8) {
    const int rb = (t / 5) * 16, ct = (t % 5) * 16;
    v8f acc = zero8();
#pragma unroll
    for (int ks = 0; ks < 2; ++ks) {
      v16bf a = load_a16(&lQ[0][0], 72, rb, ks * 32);
      v16bf bb = load_b16(&lKt[0][0], 72, ct, ks * 32);
      acc = WMMA_BF16(a, bb, acc);
    }
    const int ln = lane & 15, h8 = (lane >> 4) * 8;
#pragma unroll
    for (int r = 0; r < 8; ++r) lS[rb + h8 + r][ct + ln] = acc[r];
  }
  __syncthreads();

  // ---- branchless masked softmax per row; bf16 P written in place over lS ----
  if (tid < 64) {
    const int r = tid;
    float mx = -FLT_MAX;
    for (int m = 0; m < 77; ++m) {
      float s = lS[r][m];
      s = (lAllow[m] > 0.f) ? s : -FLT_MAX;
      lS[r][m] = s;
      mx = fmaxf(mx, s);
    }
    float sum = 0.f;
    for (int m = 0; m < 77; ++m) {
      float e = __expf(lS[r][m] - mx);
      lS[r][m] = e;
      sum += e;
    }
    const float inv = (sum > 0.f) ? 1.f / sum : 0.f;
    bf16* P = (bf16*)&lS[r][0];  // bf16 write at 2m never clobbers an unread float
    for (int m = 0; m < 77; ++m) {
      float e = lS[r][m];
      P[m] = (bf16)(e * inv);
    }
    for (int m = 77; m < 96; ++m) P[m] = (bf16)0.f;
  }
  __syncthreads();

  // ---- O = P @ V : 64x64 = 4x4 tiles, K padded to 96 ----
  for (int t = wave; t < 16; t += 8) {
    const int rb = (t >> 2) * 16, ct = (t & 3) * 16;
    v8f acc = zero8();
#pragma unroll
    for (int ks = 0; ks < 3; ++ks) {
      v16bf a = load_a16((const bf16*)&lS[0][0], 168, rb, ks * 32);
      v16bf bb = load_b16(&lVt[0][0], 104, ct, ks * 32);
      acc = WMMA_BF16(a, bb, acc);
    }
    const int ln = lane & 15, h8 = (lane >> 4) * 8;
#pragma unroll
    for (int r = 0; r < 8; ++r) {
      const size_t row = qrow0 + rb + h8 + r;
      ao[row * 512 + hc + ct + ln] = (bf16)acc[r];
    }
  }
}

// -------------------------------------------------------------------------
extern "C" void kernel_launch(void* const* d_in, const int* in_sizes, int n_in,
                              void* d_out, int out_size, void* d_ws, size_t ws_size,
                              hipStream_t stream) {
  (void)in_sizes; (void)n_in; (void)out_size; (void)ws_size;
  const float* x    = (const float*)d_in[0];
  const float* ctx  = (const float*)d_in[1];
  const unsigned char* mask = (const unsigned char*)d_in[2];
  const float* Wq   = (const float*)d_in[3];
  const float* Wk   = (const float*)d_in[4];
  const float* Wv   = (const float*)d_in[5];
  const float* Wout = (const float*)d_in[6];
  const float* bout = (const float*)d_in[7];

  const int BN = 16 * 4096;                 // 65536 query rows
  const int CM = 16 * 77;                   // 1232 context rows
  char* ws = (char*)d_ws;
  bf16* x_bf  = (bf16*)ws;  ws += (size_t)BN * 512 * 2;   // reused later as `ab`
  bf16* c_bf  = (bf16*)ws;  ws += (size_t)CM * 768 * 2;
  bf16* Wq_t  = (bf16*)ws;  ws += (size_t)512 * 512 * 2;
  bf16* Wk_t  = (bf16*)ws;  ws += (size_t)512 * 768 * 2;
  bf16* Wv_t  = (bf16*)ws;  ws += (size_t)512 * 768 * 2;
  bf16* Wo_t  = (bf16*)ws;  ws += (size_t)512 * 512 * 2;
  bf16* qb    = (bf16*)ws;  ws += (size_t)BN * 512 * 2;
  bf16* kb    = (bf16*)ws;  ws += (size_t)CM * 512 * 2;
  bf16* vb    = (bf16*)ws;  ws += (size_t)CM * 512 * 2;
  bf16* ab    = x_bf;       // x_bf is dead after the q projection

  const dim3 blk(256);
  // bf16 conversions (element counts are multiples of 4)
  f32_to_bf16_kernel<<<(unsigned)(((size_t)BN * 512 / 4 + 255) / 256), blk, 0, stream>>>(
      x, x_bf, (long long)BN * 512 / 4);
  f32_to_bf16_kernel<<<(unsigned)(((size_t)CM * 768 / 4 + 255) / 256), blk, 0, stream>>>(
      ctx, c_bf, (long long)CM * 768 / 4);
  // weight transposes to bf16 [N][K]
  transpose_w_kernel<<<dim3(16, 16), blk, 0, stream>>>(Wq, Wq_t, 512, 512);
  transpose_w_kernel<<<dim3(24, 16), blk, 0, stream>>>(Wk, Wk_t, 768, 512);
  transpose_w_kernel<<<dim3(24, 16), blk, 0, stream>>>(Wv, Wv_t, 768, 512);
  transpose_w_kernel<<<dim3(16, 16), blk, 0, stream>>>(Wout, Wo_t, 512, 512);

  // q = (x @ Wq) * D^-0.5 ; k = ctx @ Wk ; v = ctx @ Wv
  gemm_wmma_kernel<false><<<dim3(4, 512), blk, 0, stream>>>(
      x_bf, Wq_t, qb, nullptr, BN, 512, 512, 0.125f);
  gemm_wmma_kernel<false><<<dim3(4, 10), blk, 0, stream>>>(
      c_bf, Wk_t, kb, nullptr, CM, 768, 512, 1.0f);
  gemm_wmma_kernel<false><<<dim3(4, 10), blk, 0, stream>>>(
      c_bf, Wv_t, vb, nullptr, CM, 768, 512, 1.0f);
  // fused attention
  attn_kernel<<<dim3(64, 8, 16), blk, 0, stream>>>(qb, kb, vb, mask, ab);
  // out = ao @ Wout + bout (fp32)
  gemm_wmma_kernel<true><<<dim3(4, 512), blk, 0, stream>>>(
      ab, Wo_t, d_out, bout, BN, 512, 512, 1.0f);
}
